// Pointnet2MSG_55173149884613
// MI455X (gfx1250) — matrix operand order repaired and checked by reading, not compiled, not run on hardware
//
#include <hip/hip_runtime.h>
#include <math.h>

// ---------------------------------------------------------------------------
// PointNet++ MSG forward for MI455X (gfx1250, wave32, WMMA).
// All MLP layers run on v_wmma_f32_16x16x32_f16; activations chained in LDS.
// Weights are pre-converted once to zero-padded f16 (L2-resident) so each
// A-fragment is two unconditional 16B loads per lane (no divergent guards).
// ---------------------------------------------------------------------------

#define M_TILE 64            // columns (sample-points) per block tile
#define NB 4                 // batch
#define NPTS 8192            // input points

typedef __attribute__((ext_vector_type(16))) _Float16 v16h;
typedef __attribute__((ext_vector_type(8)))  _Float16 v8h;
typedef __attribute__((ext_vector_type(8)))  float    v8f;

// ---------------------------------------------------------------------------
// One MLP layer on a resident LDS tile:  Y[co x 64] = relu(g*(Wh @ X) + b)
// X : [kpad x 64] f16 in LDS (rows >= ci zero-filled)
// Wh: [co x kpad] f16 in global (zero-padded), kpad = roundup32(ci)
// A-fragment: lane = M mod 16, elems 0..7 -> K = kb+8h+e, 8..15 -> K+16.
// B-fragment: lane = N mod 16, elems 0..7 -> K = kb+8h+e, 8..15 -> K+16 (strided LDS gather).
// D layout: row = 16*cot + v + 8h, col = 16*nt + (lane&15).
// ---------------------------------------------------------------------------
__device__ __forceinline__ void mlp_layer_wmma(const _Float16* __restrict__ Xs,
                                               _Float16* __restrict__ Ys,
                                               const _Float16* __restrict__ Wh,
                                               const float* __restrict__ gma,
                                               const float* __restrict__ bta,
                                               int ci, int co)
{
  const int lane = threadIdx.x & 31;
  const int wv   = threadIdx.x >> 5;
  const int nw   = (int)(blockDim.x >> 5);
  const int h    = lane >> 4;
  const int mr   = lane & 15;
  const int cot_n = co >> 4;                 // co is a multiple of 16 here
  const int ktn   = (ci + 31) >> 5;          // K chunks of 32
  const int kpadw = ktn << 5;                // padded row stride of Wh
  const int ntot  = cot_n * (M_TILE / 16);

  for (int t = wv; t < ntot; t += nw) {
    const int cot = t >> 2;
    const int nt  = t & 3;
    v8f acc = {0.f, 0.f, 0.f, 0.f, 0.f, 0.f, 0.f, 0.f};
    const int M = cot * 16 + mr;
    const _Float16* Wrow = Wh + (size_t)M * (size_t)kpadw;

    for (int kt = 0; kt < ktn; ++kt) {
      const int kb = kt << 5;
      // next K-chunk weight prefetch -> global_prefetch_b8
      __builtin_prefetch(Wrow + kb + 32, 0, 0);
      const v8h alo = *(const v8h*)(Wrow + kb + (h << 3));
      const v8h ahi = *(const v8h*)(Wrow + kb + 16 + (h << 3));
      const v16h a = __builtin_shufflevector(alo, ahi,
                       0, 1, 2, 3, 4, 5, 6, 7, 8, 9, 10, 11, 12, 13, 14, 15);
      v16h bf;
#pragma unroll
      for (int s = 0; s < 16; ++s)
        bf[s] = Xs[(size_t)(kb + (s < 8 ? (h << 3) + s : 16 + (h << 3) + (s - 8))) * M_TILE + nt * 16 + mr];
      acc = __builtin_amdgcn_wmma_f32_16x16x32_f16(false, a, false, bf,
                                                   (short)0, acc, false, false);
    }

    const int rbase = cot * 16 + (h ? 8 : 0);
    const int col   = nt * 16 + mr;
#pragma unroll
    for (int v = 0; v < 8; ++v) {
      const int row = rbase + v;
      float y = acc[v] * gma[row] + bta[row];
      y = y > 0.f ? y : 0.f;
      Ys[(size_t)row * M_TILE + col] = (_Float16)y;
    }
  }
}

// ---------------------------------------------------------------------------
// Weight preparation: W[co x ci] f32 -> Wh[co x kpad] f16, zero padded.
// ---------------------------------------------------------------------------
__global__ void prep_w_kernel(const float* __restrict__ W, _Float16* __restrict__ Wh,
                              int co, int ci, int kpad)
{
  const int i = blockIdx.x * blockDim.x + threadIdx.x;
  if (i >= co * kpad) return;
  const int r = i / kpad, k = i % kpad;
  Wh[i] = (k < ci) ? (_Float16)W[(size_t)r * ci + k] : (_Float16)0.f;
}

// ---------------------------------------------------------------------------
// Split pointcloud (B,N,6) -> xyz (B,N,3), feat (B,3,N), and d_out xyz copy.
// ---------------------------------------------------------------------------
__global__ void split_kernel(const float* __restrict__ pc, float* __restrict__ xyz,
                             float* __restrict__ feat, float* __restrict__ out_xyz)
{
  const int i = blockIdx.x * blockDim.x + threadIdx.x;
  if (i >= NB * NPTS) return;
  const int b = i >> 13, n = i & (NPTS - 1);
#pragma unroll
  for (int k = 0; k < 3; ++k) {
    const float v = pc[(size_t)i * 6 + k];
    xyz[(size_t)i * 3 + k] = v;
    out_xyz[(size_t)i * 3 + k] = v;
    feat[((size_t)b * 3 + k) * NPTS + n] = pc[(size_t)i * 6 + 3 + k];
  }
}

// ---------------------------------------------------------------------------
// Farthest point sampling. One block (256 threads) per batch, xyz in LDS.
// Matches reference: fps[0]=0; then argmax (lowest index on tie) + min-update.
// ---------------------------------------------------------------------------
template <int PPT>
__global__ void fps_kernel(const float* __restrict__ xyz, int Nsrc, int npoint,
                           int* __restrict__ fidx, float* __restrict__ nxyz)
{
  extern __shared__ char smem[];
  float* sx = (float*)smem;                  // [Nsrc*3]
  float* rv = sx + Nsrc * 3;                 // [256]
  int*   ri = (int*)(rv + 256);              // [256]
  float* sp = (float*)(ri + 256);            // [3]
  const int b = blockIdx.x;
  const int tid = threadIdx.x;
  const float* xb = xyz + (size_t)b * Nsrc * 3;
  for (int i = tid; i < Nsrc * 3; i += 256) sx[i] = xb[i];
  __syncthreads();

  const float x0 = sx[0], y0 = sx[1], z0 = sx[2];
  float d[PPT];
#pragma unroll
  for (int i = 0; i < PPT; ++i) {
    const int p = tid * PPT + i;
    const float dx = sx[p * 3] - x0, dy = sx[p * 3 + 1] - y0, dz = sx[p * 3 + 2] - z0;
    d[i] = dx * dx + dy * dy + dz * dz;
  }
  if (tid == 0) {
    fidx[(size_t)b * npoint] = 0;
    nxyz[(size_t)b * npoint * 3 + 0] = x0;
    nxyz[(size_t)b * npoint * 3 + 1] = y0;
    nxyz[(size_t)b * npoint * 3 + 2] = z0;
  }
  for (int k = 1; k < npoint; ++k) {
    float bv = -1.f; int bi = tid * PPT;
#pragma unroll
    for (int i = 0; i < PPT; ++i)
      if (d[i] > bv) { bv = d[i]; bi = tid * PPT + i; }
    rv[tid] = bv; ri[tid] = bi;
    __syncthreads();
    for (int s = 128; s > 0; s >>= 1) {
      if (tid < s) {
        const float ov = rv[tid + s]; const int oi = ri[tid + s];
        if (ov > rv[tid] || (ov == rv[tid] && oi < ri[tid])) { rv[tid] = ov; ri[tid] = oi; }
      }
      __syncthreads();
    }
    if (tid == 0) {
      const int sel = ri[0];
      fidx[(size_t)b * npoint + k] = sel;
      sp[0] = sx[sel * 3]; sp[1] = sx[sel * 3 + 1]; sp[2] = sx[sel * 3 + 2];
      nxyz[((size_t)b * npoint + k) * 3 + 0] = sp[0];
      nxyz[((size_t)b * npoint + k) * 3 + 1] = sp[1];
      nxyz[((size_t)b * npoint + k) * 3 + 2] = sp[2];
    }
    __syncthreads();
    const float qx = sp[0], qy = sp[1], qz = sp[2];
#pragma unroll
    for (int i = 0; i < PPT; ++i) {
      const int p = tid * PPT + i;
      const float dx = sx[p * 3] - qx, dy = sx[p * 3 + 1] - qy, dz = sx[p * 3 + 2] - qz;
      const float dd = dx * dx + dy * dy + dz * dz;
      d[i] = d[i] < dd ? d[i] : dd;
    }
    __syncthreads();
  }
}

// ---------------------------------------------------------------------------
// Ball query: first `ns` indices with d2 < r2 (== top_k(-score) of reference),
// padded with the first hit.
// ---------------------------------------------------------------------------
__global__ void ball_query_kernel(const float* __restrict__ xyz,
                                  const float* __restrict__ nxyz,
                                  int Nsrc, int npoint, float r2, int ns,
                                  int* __restrict__ out)
{
  const int s = blockIdx.x * blockDim.x + threadIdx.x;
  if (s >= npoint) return;
  const int b = blockIdx.z;
  const float* q = nxyz + ((size_t)b * npoint + s) * 3;
  const float qx = q[0], qy = q[1], qz = q[2];
  const float* xb = xyz + (size_t)b * Nsrc * 3;
  int* o = out + ((size_t)b * npoint + s) * ns;
  int cnt = 0;
  for (int n = 0; n < Nsrc && cnt < ns; ++n) {
    const float dx = xb[n * 3] - qx, dy = xb[n * 3 + 1] - qy, dz = xb[n * 3 + 2] - qz;
    if (dx * dx + dy * dy + dz * dz < r2) o[cnt++] = n;
  }
  const int first = (cnt > 0) ? o[0] : 0;
  for (; cnt < ns; ++cnt) o[cnt] = first;
}

// ---------------------------------------------------------------------------
// Fused SA MLP: gather-group tile into LDS, 3 WMMA layers, max over nsample.
// ---------------------------------------------------------------------------
__global__ void sa_mlp_kernel(const int* __restrict__ balli,
                              const float* __restrict__ xyz,
                              const float* __restrict__ nxyz,
                              const float* __restrict__ feat,
                              float* __restrict__ feat_out,
                              int Nsrc, int npoint, int ns, int cin,
                              int choff, int Ctot,
                              const _Float16* W0, const float* g0, const float* b0, int c1,
                              const _Float16* W1, const float* g1, const float* b1, int c2,
                              const _Float16* W2, const float* g2, const float* b2, int c3,
                              int kpad0)
{
  extern __shared__ char smem[];
  _Float16* bufA = (_Float16*)smem;                       // [kpad0 x 64]
  _Float16* bufB = bufA + (size_t)kpad0 * M_TILE;         // [256  x 64]
  const int b = blockIdx.z;
  const int cpb = M_TILE / ns;
  const int c0 = blockIdx.x * cpb;
  const int C = cin + 3;

  // Build grouped input tile: rows 0..2 = xyz - center, rows 3..C-1 = features.
  for (int li = threadIdx.x; li < C * M_TILE; li += blockDim.x) {
    const int r = li / M_TILE, m = li % M_TILE;
    const int cl = m / ns, j = m % ns;
    const int ctr = c0 + cl;
    int p = balli[((size_t)b * npoint + ctr) * ns + j];
    if (p < 0 || p >= Nsrc) p = 0;
    float x;
    if (r < 3)
      x = xyz[((size_t)b * Nsrc + p) * 3 + r] - nxyz[((size_t)b * npoint + ctr) * 3 + r];
    else
      x = feat[((size_t)b * cin + (r - 3)) * Nsrc + p];
    bufA[li] = (_Float16)x;
  }
  for (int li = C * M_TILE + threadIdx.x; li < kpad0 * M_TILE; li += blockDim.x)
    bufA[li] = (_Float16)0.f;
  __syncthreads();

  mlp_layer_wmma(bufA, bufB, W0, g0, b0, C, c1);   __syncthreads(); // A -> B
  mlp_layer_wmma(bufB, bufA, W1, g1, b1, c1, c2);  __syncthreads(); // B -> A
  mlp_layer_wmma(bufA, bufB, W2, g2, b2, c2, c3);  __syncthreads(); // A -> B

  // Max-pool over nsample per center; ReLU'd values => init 0 is exact.
  for (int o = threadIdx.x; o < c3 * cpb; o += blockDim.x) {
    const int ch = o / cpb, cl = o % cpb;
    float m = 0.f;
    for (int j = 0; j < ns; ++j) {
      const float v = (float)bufB[(size_t)ch * M_TILE + cl * ns + j];
      m = fmaxf(m, v);
    }
    feat_out[((size_t)b * Ctot + choff + ch) * npoint + (c0 + cl)] = m;
  }
}

// ---------------------------------------------------------------------------
// 3-NN + inverse-distance weights for feature propagation.
// ---------------------------------------------------------------------------
__global__ void knn3_kernel(const float* __restrict__ unk, const float* __restrict__ kn,
                            int Nu, int Nk, int* __restrict__ oidx, float* __restrict__ ow)
{
  const int n = blockIdx.x * blockDim.x + threadIdx.x;
  if (n >= Nu) return;
  const int b = blockIdx.z;
  const float* q = unk + ((size_t)b * Nu + n) * 3;
  const float qx = q[0], qy = q[1], qz = q[2];
  const float* kb = kn + (size_t)b * Nk * 3;
  float d0 = 3.4e38f, d1 = 3.4e38f, d2 = 3.4e38f;
  int i0 = 0, i1 = 0, i2 = 0;
  for (int m = 0; m < Nk; ++m) {
    const float dx = kb[m * 3] - qx, dy = kb[m * 3 + 1] - qy, dz = kb[m * 3 + 2] - qz;
    const float dd = dx * dx + dy * dy + dz * dz;
    if (dd < d0)      { d2 = d1; i2 = i1; d1 = d0; i1 = i0; d0 = dd; i0 = m; }
    else if (dd < d1) { d2 = d1; i2 = i1; d1 = dd; i1 = m; }
    else if (dd < d2) { d2 = dd; i2 = m; }
  }
  const float w0 = 1.f / (sqrtf(fmaxf(d0, 1e-12f)) + 1e-8f);
  const float w1 = 1.f / (sqrtf(fmaxf(d1, 1e-12f)) + 1e-8f);
  const float w2 = 1.f / (sqrtf(fmaxf(d2, 1e-12f)) + 1e-8f);
  const float s = 1.f / (w0 + w1 + w2);
  const size_t o = ((size_t)b * Nu + n) * 3;
  oidx[o] = i0; oidx[o + 1] = i1; oidx[o + 2] = i2;
  ow[o] = w0 * s; ow[o + 1] = w1 * s; ow[o + 2] = w2 * s;
}

// ---------------------------------------------------------------------------
// Build FP input: X = concat(interp(kn_feats), unk_feats) as (B, Ct, Nu).
// ---------------------------------------------------------------------------
__global__ void build_xin_kernel(const float* __restrict__ knf, const float* __restrict__ unf,
                                 const int* __restrict__ kidx, const float* __restrict__ kw,
                                 int Ckn, int Cunk, int Nu, int Nk,
                                 float* __restrict__ xin)
{
  const int b = blockIdx.z;
  const int Ct = Ckn + Cunk;
  const int li = blockIdx.x * blockDim.x + threadIdx.x;
  if (li >= Ct * Nu) return;
  const int c = li / Nu, n = li % Nu;
  float v;
  if (c < Ckn) {
    const size_t o = ((size_t)b * Nu + n) * 3;
    const float* f = knf + ((size_t)b * Ckn + c) * Nk;
    v = kw[o] * f[kidx[o]] + kw[o + 1] * f[kidx[o + 1]] + kw[o + 2] * f[kidx[o + 2]];
  } else {
    v = unf[((size_t)b * Cunk + (c - Ckn)) * Nu + n];
  }
  xin[((size_t)b * Ct + c) * Nu + n] = v;
}

// ---------------------------------------------------------------------------
// FP pointwise 2-layer WMMA MLP on 64-column tiles.
// ---------------------------------------------------------------------------
__global__ void fp_mlp_kernel(const float* __restrict__ xin, float* __restrict__ out,
                              int Ct, int Nn, int kpad0,
                              const _Float16* W0, const float* g0, const float* b0, int c1,
                              const _Float16* W1, const float* g1, const float* b1, int c2)
{
  extern __shared__ char smem[];
  _Float16* bufA = (_Float16*)smem;
  _Float16* bufB = bufA + (size_t)kpad0 * M_TILE;
  const int b = blockIdx.z;
  const int n0 = blockIdx.x * M_TILE;

  for (int li = threadIdx.x; li < Ct * M_TILE; li += blockDim.x) {
    const int r = li / M_TILE, m = li % M_TILE;
    bufA[li] = (_Float16)xin[((size_t)b * Ct + r) * Nn + n0 + m];
  }
  for (int li = Ct * M_TILE + threadIdx.x; li < kpad0 * M_TILE; li += blockDim.x)
    bufA[li] = (_Float16)0.f;
  __syncthreads();

  mlp_layer_wmma(bufA, bufB, W0, g0, b0, Ct, c1);  __syncthreads(); // A -> B
  mlp_layer_wmma(bufB, bufA, W1, g1, b1, c1, c2);  __syncthreads(); // B -> A

  for (int o = threadIdx.x; o < c2 * M_TILE; o += blockDim.x) {
    const int ch = o / M_TILE, col = o % M_TILE;
    out[((size_t)b * c2 + ch) * Nn + n0 + col] = (float)bufA[(size_t)ch * M_TILE + col];
  }
}

// ---------------------------------------------------------------------------
// Host orchestration
// ---------------------------------------------------------------------------
static void run_sa_scale(const float* xyz, const float* nxyz, const float* feat,
                         float* feat_out, const int* balli,
                         int Nsrc, int npoint, int ns, int cin, int choff, int Ctot,
                         const _Float16* W0, const float* g0, const float* b0, int c1,
                         const _Float16* W1, const float* g1, const float* b1, int c2,
                         const _Float16* W2, const float* g2, const float* b2, int c3,
                         hipStream_t stream)
{
  const int C = cin + 3;
  const int kpad = ((C + 31) / 32) * 32;
  const int cpb = M_TILE / ns;
  const size_t sh = (size_t)(kpad + 256) * M_TILE * 2;
  sa_mlp_kernel<<<dim3(npoint / cpb, 1, NB), 256, sh, stream>>>(
      balli, xyz, nxyz, feat, feat_out, Nsrc, npoint, ns, cin, choff, Ctot,
      W0, g0, b0, c1, W1, g1, b1, c2, W2, g2, b2, c3, kpad);
}

static void run_fp(const float* unk_xyz, const float* kn_xyz,
                   const float* kn_feat, const float* unk_feat,
                   int Nu, int Nk, int Ckn, int Cunk,
                   int* knni, float* knnw, float* xin, float* outf,
                   const _Float16* W0, const float* g0, const float* b0, int c1,
                   const _Float16* W1, const float* g1, const float* b1, int c2,
                   hipStream_t stream)
{
  knn3_kernel<<<dim3((Nu + 127) / 128, 1, NB), 128, 0, stream>>>(
      unk_xyz, kn_xyz, Nu, Nk, knni, knnw);
  const int Ct = Ckn + Cunk;
  build_xin_kernel<<<dim3((Ct * Nu + 255) / 256, 1, NB), 256, 0, stream>>>(
      kn_feat, unk_feat, knni, knnw, Ckn, Cunk, Nu, Nk, xin);
  const int kpad = ((Ct + 31) / 32) * 32;
  const size_t sh = (size_t)(kpad + 256) * M_TILE * 2;
  fp_mlp_kernel<<<dim3(Nu / M_TILE, 1, NB), 256, sh, stream>>>(
      xin, outf, Ct, Nu, kpad, W0, g0, b0, c1, W1, g1, b1, c2);
}

extern "C" void kernel_launch(void* const* d_in, const int* in_sizes, int n_in,
                              void* d_out, int out_size, void* d_ws, size_t ws_size,
                              hipStream_t stream)
{
  (void)in_sizes; (void)n_in; (void)out_size; (void)ws_size;
#define DF(i) ((const float*)d_in[(i)])
  const float* pc = DF(0);
  float* ws  = (float*)d_ws;
  float* out = (float*)d_out;

  // workspace layout (float elements)
  float* xyz0   = ws + 0;         // 4*8192*3
  float* feat0  = ws + 98304;     // 4*3*8192
  float* xyz1   = ws + 196608;    // 4*2048*3
  float* xyz2   = ws + 221184;    // 4*512*3
  float* xyz3   = ws + 227328;    // 4*128*3
  float* feat1  = ws + 228864;    // 4*128*2048
  float* feat2  = ws + 1277440;   // 4*256*512
  float* feat3  = ws + 1801728;   // 4*512*128
  float* feat2n = ws + 2063872;   // 4*256*512
  float* feat1n = ws + 2588160;   // 4*128*2048
  int*   fpsi   = (int*)(ws + 3636736);   // 4*2048
  int*   balli  = (int*)(ws + 3644928);   // 4*2048*32
  int*   knni   = (int*)(ws + 3907072);   // 4*8192*3
  float* knnw   = ws + 4005376;           // 4*8192*3
  float* xin    = ws + 4103680;           // max 4*131*8192 = 4292608
  _Float16* whbase = (_Float16*)(ws + 8396288); // padded f16 weights, ~1.3MB

  // --- weight prep: f32 (co x ci) -> f16 (co x kpad), zero padded ---
  static const int wdi[24] = { 1, 4, 7,  10, 13, 16,  19, 22, 25,  28, 31, 34,
                               37, 40, 43, 46, 49, 52, 55, 58, 61, 64, 67, 70 };
  static const int wco[24] = { 32, 32, 64, 32, 32, 64, 64, 64, 128, 64, 64, 128,
                               128, 128, 256, 128, 128, 256, 128, 128, 256, 128, 256, 256 };
  static const int wci[24] = { 6, 32, 32, 6, 32, 32, 131, 64, 64, 131, 64, 64,
                               259, 128, 128, 259, 128, 128, 131, 128, 384, 256, 768, 256 };
  const _Float16* wh[24];
  {
    _Float16* cur = whbase;
    for (int i = 0; i < 24; ++i) {
      const int kpad = ((wci[i] + 31) / 32) * 32;
      const int tot = wco[i] * kpad;
      wh[i] = cur;
      prep_w_kernel<<<dim3((tot + 255) / 256), 256, 0, stream>>>(
          DF(wdi[i]), cur, wco[i], wci[i], kpad);
      cur += tot;
    }
  }

  // --- split input ---
  split_kernel<<<dim3((NB * NPTS + 255) / 256), 256, 0, stream>>>(pc, xyz0, feat0, out);

  // --- SA stage 0: 8192 -> 2048, cin=3, layers [32,32,64], Ctot=128 ---
  fps_kernel<32><<<dim3(NB), 256, NPTS * 12 + 256 * 8 + 16, stream>>>(xyz0, 8192, 2048, fpsi, xyz1);
  ball_query_kernel<<<dim3(16, 1, NB), 128, 0, stream>>>(xyz0, xyz1, 8192, 2048, 0.01f, 16, balli);
  run_sa_scale(xyz0, xyz1, feat0, feat1, balli, 8192, 2048, 16, 3, 0, 128,
               wh[0], DF(2), DF(3), 32, wh[1], DF(5), DF(6), 32, wh[2], DF(8), DF(9), 64, stream);
  ball_query_kernel<<<dim3(16, 1, NB), 128, 0, stream>>>(xyz0, xyz1, 8192, 2048, 0.04f, 32, balli);
  run_sa_scale(xyz0, xyz1, feat0, feat1, balli, 8192, 2048, 32, 3, 64, 128,
               wh[3], DF(11), DF(12), 32, wh[4], DF(14), DF(15), 32, wh[5], DF(17), DF(18), 64, stream);

  // --- SA stage 1: 2048 -> 512, cin=128, layers [64,64,128], Ctot=256 ---
  fps_kernel<8><<<dim3(NB), 256, 2048 * 12 + 256 * 8 + 16, stream>>>(xyz1, 2048, 512, fpsi, xyz2);
  ball_query_kernel<<<dim3(4, 1, NB), 128, 0, stream>>>(xyz1, xyz2, 2048, 512, 0.04f, 16, balli);
  run_sa_scale(xyz1, xyz2, feat1, feat2, balli, 2048, 512, 16, 128, 0, 256,
               wh[6], DF(20), DF(21), 64, wh[7], DF(23), DF(24), 64, wh[8], DF(26), DF(27), 128, stream);
  ball_query_kernel<<<dim3(4, 1, NB), 128, 0, stream>>>(xyz1, xyz2, 2048, 512, 0.16f, 32, balli);
  run_sa_scale(xyz1, xyz2, feat1, feat2, balli, 2048, 512, 32, 128, 128, 256,
               wh[9], DF(29), DF(30), 64, wh[10], DF(32), DF(33), 64, wh[11], DF(35), DF(36), 128, stream);

  // --- SA stage 2: 512 -> 128, cin=256, layers [128,128,256], Ctot=512 ---
  fps_kernel<2><<<dim3(NB), 256, 512 * 12 + 256 * 8 + 16, stream>>>(xyz2, 512, 128, fpsi, xyz3);
  ball_query_kernel<<<dim3(1, 1, NB), 128, 0, stream>>>(xyz2, xyz3, 512, 128, 0.16f, 16, balli);
  run_sa_scale(xyz2, xyz3, feat2, feat3, balli, 512, 128, 16, 256, 0, 512,
               wh[12], DF(38), DF(39), 128, wh[13], DF(41), DF(42), 128, wh[14], DF(44), DF(45), 256, stream);
  ball_query_kernel<<<dim3(1, 1, NB), 128, 0, stream>>>(xyz2, xyz3, 512, 128, 0.64f, 32, balli);
  run_sa_scale(xyz2, xyz3, feat2, feat3, balli, 512, 128, 32, 256, 256, 512,
               wh[15], DF(47), DF(48), 128, wh[16], DF(50), DF(51), 128, wh[17], DF(53), DF(54), 256, stream);

  // --- FP stage 2: (512 unk, 128 known), C=512+256=768 -> [256,256] ---
  run_fp(xyz2, xyz3, feat3, feat2, 512, 128, 512, 256, knni, knnw, xin, feat2n,
         wh[22], DF(68), DF(69), 256, wh[23], DF(71), DF(72), 256, stream);

  // --- FP stage 1: (2048 unk, 512 known), C=256+128=384 -> [256,128] ---
  run_fp(xyz1, xyz2, feat2n, feat1, 2048, 512, 256, 128, knni, knnw, xin, feat1n,
         wh[20], DF(62), DF(63), 256, wh[21], DF(65), DF(66), 128, stream);

  // --- FP stage 0: (8192 unk, 2048 known), C=128+3=131 -> [128,128] -> d_out ---
  run_fp(xyz0, xyz1, feat1n, feat0, 8192, 2048, 128, 3, knni, knnw, xin, out + 98304,
         wh[18], DF(56), DF(57), 128, wh[19], DF(59), DF(60), 128, stream);
#undef DF
}
